// MultiLevelLoss_3307124818295
// MI455X (gfx1250) — matrix operand order repaired and checked by reading, not compiled, not run on hardware
//
#include <hip/hip_runtime.h>
#include <cstdint>

typedef __bf16 bf16_t;
typedef __attribute__((ext_vector_type(16))) __bf16 v16bf;
typedef __attribute__((ext_vector_type(8)))  float  v8f;

// ---------------------------------------------------------------------------
// Workspace layout:
//   [0 .. 2MB)   : bf16 rep matrix (8192 x 128 max), reused for both levels
//   [2MB .. )    : float accumulators:
//       acc[0] = seg numerator   acc[1] = seg denominator (sum of weights)
//       acc[2] = regional  sum(logz) - sum(pos)
//       acc[3] = pixel     sum(logz) - sum(pos)
//       acc[4] = global    sum(logz - pos)
// ---------------------------------------------------------------------------
#define REP_BYTES (8192u * 128u * 2u)

__global__ void k_init(float* acc) {
    if (threadIdx.x < 8) acc[threadIdx.x] = 0.0f;
}

// ------------------------- weighted seg cross-entropy ----------------------
// logits (4,2,256,256), mask (4,256,256), weight (2)
__global__ void k_seg(const float* __restrict__ logits,
                      const int*  __restrict__ mask,
                      const float* __restrict__ weight,
                      float* acc) {
    int p  = blockIdx.x * 256 + threadIdx.x;          // 0 .. 262143
    int b  = p >> 16;
    int hw = p & 65535;
    const float* lp = logits + ((size_t)b << 17);
    float x0 = lp[hw];
    float x1 = lp[65536 + hw];
    int   m  = mask[p];
    float mx  = fmaxf(x0, x1);
    float lse = mx + __logf(__expf(x0 - mx) + __expf(x1 - mx));
    float xm  = m ? x1 : x0;
    float w   = weight[m];
    float num = w * (lse - xm);
    #pragma unroll
    for (int off = 16; off; off >>= 1) {
        num += __shfl_xor(num, off, 32);
        w   += __shfl_xor(w,   off, 32);
    }
    if ((threadIdx.x & 31) == 0) {
        atomicAdd(&acc[0], num);
        atomicAdd(&acc[1], w);
    }
}

// ------------------------- global NT-Xent (N=4, D=256) ---------------------
__global__ void k_global(const float* __restrict__ g1,
                         const float* __restrict__ g2,
                         float* accG) {
    __shared__ float z[8][256];
    int lane = threadIdx.x & 31;
    int w    = threadIdx.x >> 5;                      // wave = row (0..7)
    const float* src = (w < 4) ? (g1 + w * 256) : (g2 + (w - 4) * 256);
    float v[8];
    float ss = 0.0f;
    #pragma unroll
    for (int j = 0; j < 8; ++j) { v[j] = src[lane + 32 * j]; ss += v[j] * v[j]; }
    #pragma unroll
    for (int off = 16; off; off >>= 1) ss += __shfl_xor(ss, off, 32);
    float inv = 1.0f / fmaxf(sqrtf(ss), 1e-12f);
    #pragma unroll
    for (int j = 0; j < 8; ++j) z[w][lane + 32 * j] = v[j] * inv;
    __syncthreads();
    float dot[8];
    #pragma unroll
    for (int c = 0; c < 8; ++c) {
        float pth = 0.0f;
        #pragma unroll
        for (int j = 0; j < 8; ++j) pth += z[w][lane + 32 * j] * z[c][lane + 32 * j];
        #pragma unroll
        for (int off = 16; off; off >>= 1) pth += __shfl_xor(pth, off, 32);
        dot[c] = pth;
    }
    if (lane == 0) {
        // unit rows: max(sim/temp) == 10 exactly on the (unmasked) diagonal
        float s = 0.0f;
        #pragma unroll
        for (int c = 0; c < 8; ++c) s += __expf(dot[c] * 10.0f - 10.0f);
        float logz = 10.0f + __logf(s);
        float pos  = dot[(w + 4) & 7] * 10.0f;
        atomicAdd(accG, logz - pos);
    }
}

// ---------------- normalized bf16 rep rows: regional level -----------------
// r1,r2: (4,128,32,32) -> rows (b*1024+s), d stride 1024
__global__ void k_build_rep_r(const float* __restrict__ r1,
                              const float* __restrict__ r2,
                              bf16_t* __restrict__ rep) {
    int lane = threadIdx.x & 31;
    int row  = blockIdx.x * 8 + (threadIdx.x >> 5);   // 0..8191
    const float* src = (row < 4096) ? r1 : r2;
    int rr = row & 4095;
    int b  = rr >> 10;
    int s  = rr & 1023;
    const float* base = src + (size_t)b * 131072 + s;
    float v[4];
    float ss = 0.0f;
    #pragma unroll
    for (int j = 0; j < 4; ++j) {
        int d = lane + 32 * j;
        v[j] = base[(size_t)d * 1024];
        ss += v[j] * v[j];
    }
    #pragma unroll
    for (int off = 16; off; off >>= 1) ss += __shfl_xor(ss, off, 32);
    float inv = 1.0f / fmaxf(sqrtf(ss), 1e-12f);
    bf16_t* out = rep + (size_t)row * 128;
    #pragma unroll
    for (int j = 0; j < 4; ++j) out[lane + 32 * j] = (bf16_t)(v[j] * inv);
}

// ---------------- normalized bf16 rep rows: pixel level --------------------
// p1,p2: (4,64,256,256); 4x4 mean pool -> (4,64,64,64); gather idx; normalize
__global__ void k_build_rep_p(const float* __restrict__ p1,
                              const float* __restrict__ p2,
                              const int* __restrict__ idx,
                              bf16_t* __restrict__ rep) {
    int lane = threadIdx.x & 31;
    int row  = blockIdx.x * 8 + (threadIdx.x >> 5);   // 0..8191
    const float* src = (row < 4096) ? p1 : p2;
    int rr = row & 4095;
    int b  = rr >> 10;
    int s  = idx[rr & 1023];
    int sy = s >> 6, sx = s & 63;
    float v[2];
    float ss = 0.0f;
    #pragma unroll
    for (int j = 0; j < 2; ++j) {
        int d = lane + 32 * j;
        const float* base = src + (((size_t)(b * 64 + d) * 256 + sy * 4) * 256) + sx * 4;
        float a = 0.0f;
        #pragma unroll
        for (int y = 0; y < 4; ++y)
            #pragma unroll
            for (int x = 0; x < 4; ++x) a += base[y * 256 + x];
        v[j] = a * (1.0f / 16.0f);
        ss += v[j] * v[j];
    }
    #pragma unroll
    for (int off = 16; off; off >>= 1) ss += __shfl_xor(ss, off, 32);
    float inv = 1.0f / fmaxf(sqrtf(ss), 1e-12f);
    bf16_t* out = rep + (size_t)row * 64;
    #pragma unroll
    for (int j = 0; j < 2; ++j) out[lane + 32 * j] = (bf16_t)(v[j] * inv);
}

// ------------------------- positive-pair terms -----------------------------
template <int D>
__global__ void k_pos(const bf16_t* __restrict__ rep, float* acc) {
    int lane = threadIdx.x & 31;
    int row  = blockIdx.x * 8 + (threadIdx.x >> 5);
    int pair = (row + 4096) & 8191;
    const bf16_t* a = rep + (size_t)row  * D;
    const bf16_t* b = rep + (size_t)pair * D;
    float dot = 0.0f;
    #pragma unroll
    for (int j = 0; j < D / 32; ++j) {
        int d = lane + 32 * j;
        dot += (float)a[d] * (float)b[d];
    }
    #pragma unroll
    for (int off = 16; off; off >>= 1) dot += __shfl_xor(dot, off, 32);
    if (lane == 0) atomicAdd(acc, -dot * 10.0f);
}

// ---------------- WMMA similarity GEMM + streaming logsumexp ---------------
// One block = one 16-row tile; 8 waves stride over the 512 16-col tiles.
// A frag (16x32 bf16): lanes 0-15 rows, K 0-15; lanes
// 16-31 map K 16-31.  B frag (32x16): lane = column, halves = K run,
// lanes 16-31 hold K+16 -> one contiguous 32B global load per lane per chunk.
// B operand is double-buffered (ping-pong, 2x unrolled column loop) so the
// load clause for tile i+1 overlaps the WMMA chain + exp work of tile i.
template <int D>
__global__ void __launch_bounds__(256) k_gemm(const bf16_t* __restrict__ rep,
                                              float* acc) {
    constexpr int KC = D / 32;
    __shared__ bf16_t tileA[16 * D];
    __shared__ float  rowsum[8][16];
    int tid  = threadIdx.x;
    int lane = tid & 31;
    int wid  = tid >> 5;
    int m0   = blockIdx.x * 16;

    // rows m0..m0+15 are contiguous in rep
    for (int t = tid; t < 16 * D; t += 256) tileA[t] = rep[(size_t)m0 * D + t];
    __syncthreads();

    int mrow = lane & 15;
    int hi   = lane >> 4;                    // 0: lanes 0-15, 1: lanes 16-31

    v16bf afrag[KC];
    #pragma unroll
    for (int kc = 0; kc < KC; ++kc) {
        const bf16_t* basep = &tileA[mrow * D + kc * 32 + hi * 16];
        v16bf a;
        #pragma unroll
        for (int h = 0; h < 8; ++h) { a[h] = basep[h]; a[h + 8] = basep[h + 8]; }
        afrag[kc] = a;
    }

    float accRow[8] = {0.f, 0.f, 0.f, 0.f, 0.f, 0.f, 0.f, 0.f};

    // ---- pipelined column sweep: 64 tiles per wave, ping-pong B buffers ----
    auto loadB = [&](int it, v16bf* bfr) {
        int ct = wid + it * 8;
        const bf16_t* brow = rep + (size_t)(ct * 16 + mrow) * D + hi * 16;
        #pragma unroll
        for (int kc = 0; kc < KC; ++kc)
            bfr[kc] = *(const v16bf*)(brow + kc * 32);
    };
    auto tileStep = [&](const v16bf* bfr) {
        v8f c = {};
        #pragma unroll
        for (int kc = 0; kc < KC; ++kc)
            c = __builtin_amdgcn_wmma_f32_16x16x32_bf16(
                    false, afrag[kc], false, bfr[kc], (short)0, c, false, false);
        #pragma unroll
        for (int v = 0; v < 8; ++v)
            accRow[v] += __expf(c[v] * 10.0f - 10.0f);   // exp(sim/temp - 10)
    };

    v16bf b0[KC], b1[KC];
    loadB(0, b0);
    #pragma unroll 1
    for (int i = 0; i < 62; i += 2) {
        loadB(i + 1, b1);
        tileStep(b0);
        loadB(i + 2, b0);
        tileStep(b1);
    }
    loadB(63, b1);                  // b0 currently holds tile 62
    tileStep(b0);
    tileStep(b1);

    // reduce within each 16-lane half (rows 0-7 in lanes 0-15, 8-15 in 16-31)
    #pragma unroll
    for (int v = 0; v < 8; ++v)
        #pragma unroll
        for (int off = 8; off; off >>= 1)
            accRow[v] += __shfl_xor(accRow[v], off, 32);

    if (lane == 0)
        for (int v = 0; v < 8; ++v) rowsum[wid][v] = accRow[v];
    if (lane == 16)
        for (int v = 0; v < 8; ++v) rowsum[wid][8 + v] = accRow[v];
    __syncthreads();

    if (tid < 16) {
        float s = 0.0f;
        #pragma unroll
        for (int w = 0; w < 8; ++w) s += rowsum[w][tid];
        atomicAdd(acc, 10.0f + __logf(s));             // logz for this row
    }
}

// ------------------------- finalize ----------------------------------------
__global__ void k_final(const float* acc, float* out) {
    if (threadIdx.x == 0) {
        float seg = acc[0] / acc[1];
        float lr  = acc[2] * (1.0f / 8192.0f);
        float lp  = acc[3] * (1.0f / 8192.0f);
        float lg  = acc[4] * (1.0f / 8.0f);
        float total = seg + 1.0f * lp + 0.5f * lr + 0.5f * lg;
        out[0] = total; out[1] = seg; out[2] = lp; out[3] = lr; out[4] = lg;
    }
}

extern "C" void kernel_launch(void* const* d_in, const int* in_sizes, int n_in,
                              void* d_out, int out_size, void* d_ws, size_t ws_size,
                              hipStream_t stream) {
    (void)in_sizes; (void)n_in; (void)out_size; (void)ws_size;
    const float* logits = (const float*)d_in[0];
    const int*   mask   = (const int*)  d_in[1];
    const float* weight = (const float*)d_in[2];
    const float* g1     = (const float*)d_in[3];
    const float* g2     = (const float*)d_in[4];
    const float* r1     = (const float*)d_in[5];
    const float* r2     = (const float*)d_in[6];
    const float* p1     = (const float*)d_in[7];
    const float* p2     = (const float*)d_in[8];
    const int*   idx    = (const int*)  d_in[9];

    bf16_t* rep = (bf16_t*)d_ws;
    float*  acc = (float*)((char*)d_ws + REP_BYTES);
    float*  out = (float*)d_out;

    k_init<<<1, 32, 0, stream>>>(acc);
    k_seg<<<1024, 256, 0, stream>>>(logits, mask, weight, acc);
    k_global<<<1, 256, 0, stream>>>(g1, g2, &acc[4]);

    // regional level (D=128)
    k_build_rep_r<<<1024, 256, 0, stream>>>(r1, r2, rep);
    k_pos<128><<<1024, 256, 0, stream>>>(rep, &acc[2]);
    k_gemm<128><<<512, 256, 0, stream>>>(rep, &acc[2]);

    // pixel level (D=64), reuse rep buffer (stream-serialized)
    k_build_rep_p<<<1024, 256, 0, stream>>>(p1, p2, idx, rep);
    k_pos<64><<<1024, 256, 0, stream>>>(rep, &acc[3]);
    k_gemm<64><<<512, 256, 0, stream>>>(rep, &acc[3]);

    k_final<<<1, 32, 0, stream>>>(acc, out);
}
